// QLoRABigNet_72249939853571
// MI455X (gfx1250) — matrix operand order repaired and checked by reading, not compiled, not run on hardware
//
#include <hip/hip_runtime.h>

#define D_DIM 1024
#define R_DIM 32
#define G_DIM 16
#define NBLK 6
#define NLIN 3
#define ALPHA_F 0.0937f
#define LDH 40   // LDS row stride in halves (multiple of 8 -> 16B-aligned chunks)

typedef _Float16 half_t;
typedef __attribute__((ext_vector_type(16))) _Float16 v16h;
typedef __attribute__((ext_vector_type(8)))  _Float16 v8h;
typedef __attribute__((ext_vector_type(4)))  _Float16 v4h;
typedef __attribute__((ext_vector_type(8)))  float    v8f;

static __device__ __forceinline__ v8f wmma_f16(v16h a, v16h b, v8f c) {
  // D = A(16x32) * B(32x16) + C, f32 accumulate
  return __builtin_amdgcn_wmma_f32_16x16x32_f16(false, a, false, b, (short)0, c,
                                                false, false);
}

// 32-bit LDS byte offset of a generic pointer into __shared__ memory
// (addrspace(3) pointers are 32-bit LDS offsets on gfx1250).
static __device__ __forceinline__ unsigned lds_off(const void* p) {
  return (unsigned)(unsigned long long)(__attribute__((address_space(3))) const void*)p;
}

// CDNA5 async copy: LDS[wave_lds_base + dst] = MEM[src .. src+15], ASYNCcnt++
static __device__ __forceinline__ void async_cp_b128(unsigned dst, const void* src) {
  asm volatile("global_load_async_to_lds_b128 %0, %1, off"
               :: "v"(dst), "v"(src) : "memory");
}

static __device__ __forceinline__ void wait_asynccnt0() {
  asm volatile("s_wait_asynccnt 0x0" ::: "memory");
}

// A-fragment (16x32 f16): lane m=lane&15, hs=lane>>4 holds
// K = hs*16 + 0..15 of row m (same k order as frag_b).
static __device__ __forceinline__ v16h frag_a(const half_t* base, int lane) {
  const int m = lane & 15, hs = lane >> 4;
  const half_t* p = base + m * LDH;
  v8h lo = *(const v8h*)(p + hs * 16);
  v8h hi = *(const v8h*)(p + hs * 16 + 8);
  v16h r;
#pragma unroll
  for (int e = 0; e < 8; ++e) { r[e] = lo[e]; r[8 + e] = hi[e]; }
  return r;
}

// B-fragment (32x16 f16), contiguous k order (as frag_a): lane n=lane&15, ks=lane>>4 holds
// K = ks*16 + 0..15 of column n; LDS tile stores W rows row-major -> 16
// contiguous halves of row n.
static __device__ __forceinline__ v16h frag_b(const half_t* base, int lane) {
  const int n = lane & 15, ks = lane >> 4;
  const half_t* p = base + n * LDH + ks * 16;
  v8h lo = *(const v8h*)(p);
  v8h hi = *(const v8h*)(p + 8);
  v16h r;
#pragma unroll
  for (int e = 0; e < 8; ++e) { r[e] = lo[e]; r[8 + e] = hi[e]; }
  return r;
}

// ---------------------------------------------------------------------------
// h = x ; act = f16(x)
__global__ __launch_bounds__(256) void k_init(const float* __restrict__ x,
                                              float* __restrict__ h,
                                              half_t* __restrict__ act) {
  const size_t i = (size_t)blockIdx.x * 256 + threadIdx.x;  // float4 index
  float4 v = ((const float4*)x)[i];
  ((float4*)h)[i] = v;
  v4h a;
  a[0] = (half_t)v.x; a[1] = (half_t)v.y; a[2] = (half_t)v.z; a[3] = (half_t)v.w;
  ((v4h*)act)[i] = a;
}

// ---------------------------------------------------------------------------
// t[m, r] = ALPHA * sum_k act[m,k] * la[r,k]   (output f16 [1024,32])
// Block: 128 rows of act, 8 waves, each wave one 16x32 output tile (2 WMMA).
// act tile goes global->LDS via async DMA; la slice is cvt'ed through VALU.
__global__ __launch_bounds__(256) void k_lora_t(const half_t* __restrict__ act,
                                                const float* __restrict__ la,
                                                half_t* __restrict__ tout) {
  __shared__ half_t As[128 * LDH];
  __shared__ half_t Ls[32 * LDH];
  const int tid = threadIdx.x, lane = tid & 31, wave = tid >> 5;
  const int bm = blockIdx.x * 128;
  const int arow = tid >> 1, ako = (tid & 1) * 16;

  v8f acc[2];
#pragma unroll
  for (int s = 0; s < 2; ++s)
#pragma unroll
    for (int e = 0; e < 8; ++e) acc[s][e] = 0.0f;

#pragma unroll 1
  for (int kt = 0; kt < 32; ++kt) {
    const int ka = kt * 32 + ako;
    // async: activation tile straight into LDS
    const half_t* ap = act + (size_t)(bm + arow) * D_DIM + ka;
    const unsigned ad = lds_off(&As[arow * LDH + ako]);
    async_cp_b128(ad, ap);
    async_cp_b128(ad + 16, ap + 8);
    if (tid < 64) {  // arow in [0,32): stage la slice as f16
      const float* lp = la + (size_t)arow * D_DIM + ka;
      float4 f0 = ((const float4*)lp)[0];
      float4 f1 = ((const float4*)lp)[1];
      float4 f2 = ((const float4*)lp)[2];
      float4 f3 = ((const float4*)lp)[3];
      half_t* q = &Ls[arow * LDH + ako];
      q[0]  = (half_t)f0.x; q[1]  = (half_t)f0.y; q[2]  = (half_t)f0.z; q[3]  = (half_t)f0.w;
      q[4]  = (half_t)f1.x; q[5]  = (half_t)f1.y; q[6]  = (half_t)f1.z; q[7]  = (half_t)f1.w;
      q[8]  = (half_t)f2.x; q[9]  = (half_t)f2.y; q[10] = (half_t)f2.z; q[11] = (half_t)f2.w;
      q[12] = (half_t)f3.x; q[13] = (half_t)f3.y; q[14] = (half_t)f3.z; q[15] = (half_t)f3.w;
    }
    wait_asynccnt0();
    __syncthreads();
    v16h af = frag_a(&As[(wave * 16) * LDH], lane);
    acc[0] = wmma_f16(af, frag_b(&Ls[0], lane), acc[0]);
    acc[1] = wmma_f16(af, frag_b(&Ls[16 * LDH], lane), acc[1]);
    __syncthreads();
  }

  const int nlo = lane & 15, hs = lane >> 4;
#pragma unroll
  for (int sn = 0; sn < 2; ++sn)
#pragma unroll
    for (int r = 0; r < 8; ++r) {
      const int m = bm + wave * 16 + r + 8 * hs;
      tout[(size_t)m * R_DIM + sn * 16 + nlo] = (half_t)(ALPHA_F * acc[sn][r]);
    }
}

// ---------------------------------------------------------------------------
// out[m,n] = relu?( act @ dequant(W)^T + bias + t @ lb^T )
// Tile M=128, N=64; K-loop of 33 steps (step 32 = rank-32 LoRA chunk).
// 8 waves (4x2), each wave a 32x32 patch = 2x2 WMMA tiles.
// Double-buffered LDS: tile k+1 is fetched (A: async DMA, B: dequant via
// VALU) while WMMA consumes tile k. One barrier per K-step.
__global__ __launch_bounds__(256) void k_gemm(
    const half_t* __restrict__ act, const int* __restrict__ codes,
    const float* __restrict__ scales, const float* __restrict__ bias,
    const float* __restrict__ lb, const half_t* __restrict__ tlo,
    float* __restrict__ out, half_t* __restrict__ act_out, int relu) {
  __shared__ half_t As[2][128 * LDH];
  __shared__ half_t Bs[2][64 * LDH];
  const int tid = threadIdx.x;
  const int lane = tid & 31;
  const int wave = tid >> 5;
  const int wm = wave & 3;   // 0..3 -> rows wm*32
  const int wn = wave >> 2;  // 0..1 -> cols wn*32
  const int bm = blockIdx.y * 128;
  const int bn = blockIdx.x * 64;

  const int arow = tid >> 1, ako = (tid & 1) * 16;  // 2 thr/row, 16 halves
  const int brow = tid >> 2, bko = (tid & 3) * 8;   // 4 thr/row, 8 halves

  v8f acc[2][2];
#pragma unroll
  for (int i = 0; i < 2; ++i)
#pragma unroll
    for (int j = 0; j < 2; ++j)
#pragma unroll
      for (int e = 0; e < 8; ++e) acc[i][j][e] = 0.0f;

  // ---- prologue: stage tile 0 into buffer 0
  {
    const half_t* ap = act + (size_t)(bm + arow) * D_DIM + ako;
    const unsigned ad = lds_off(&As[0][arow * LDH + ako]);
    async_cp_b128(ad, ap);
    async_cp_b128(ad + 16, ap + 8);
    const int* cp = codes + (size_t)(bn + brow) * D_DIM + bko;
    const float s = scales[(size_t)(bn + brow) * (D_DIM / G_DIM) + (bko >> 4)];
    int4 c0 = ((const int4*)cp)[0];
    int4 c1 = ((const int4*)cp)[1];
    half_t* bp = &Bs[0][brow * LDH + bko];
    bp[0] = (half_t)((float)c0.x * s); bp[1] = (half_t)((float)c0.y * s);
    bp[2] = (half_t)((float)c0.z * s); bp[3] = (half_t)((float)c0.w * s);
    bp[4] = (half_t)((float)c1.x * s); bp[5] = (half_t)((float)c1.y * s);
    bp[6] = (half_t)((float)c1.z * s); bp[7] = (half_t)((float)c1.w * s);
    wait_asynccnt0();
    __syncthreads();
  }

#pragma unroll 1
  for (int kt = 0; kt <= 32; ++kt) {
    const int cur = kt & 1, nxt = cur ^ 1;
    const int knext = kt + 1;
    const bool have_next = (knext <= 32);

    int4 pc0 = {}, pc1 = {};
    float4 pf0 = {}, pf1 = {};
    float ps = 0.0f;
    if (have_next) {
      // A-tile of k+1: async DMA into the other LDS buffer
      const half_t* asrc = (knext < 32)
          ? act + (size_t)(bm + arow) * D_DIM + knext * 32 + ako
          : tlo + (size_t)(bm + arow) * R_DIM + ako;   // LoRA chunk: ALPHA*t
      const unsigned ad = lds_off(&As[nxt][arow * LDH + ako]);
      async_cp_b128(ad, asrc);
      async_cp_b128(ad + 16, asrc + 8);
      // B-tile of k+1 into registers (needs VALU dequant/convert)
      if (knext < 32) {
        const int kb = knext * 32 + bko;
        const int* cp = codes + (size_t)(bn + brow) * D_DIM + kb;
        pc0 = ((const int4*)cp)[0];
        pc1 = ((const int4*)cp)[1];
        ps = scales[(size_t)(bn + brow) * (D_DIM / G_DIM) + (kb >> 4)];
        if (knext < 31) __builtin_prefetch(cp + 32, 0, 1);  // k+2 codes
      } else {
        const float* lp = lb + (size_t)(bn + brow) * R_DIM + bko;
        pf0 = ((const float4*)lp)[0];
        pf1 = ((const float4*)lp)[1];
      }
    }

    // ---- compute on current buffer (overlaps the in-flight loads above)
    v16h af[2], bf[2];
#pragma unroll
    for (int sm = 0; sm < 2; ++sm)
      af[sm] = frag_a(&As[cur][(wm * 32 + sm * 16) * LDH], lane);
#pragma unroll
    for (int sn = 0; sn < 2; ++sn)
      bf[sn] = frag_b(&Bs[cur][(wn * 32 + sn * 16) * LDH], lane);
#pragma unroll
    for (int sm = 0; sm < 2; ++sm)
#pragma unroll
      for (int sn = 0; sn < 2; ++sn)
        acc[sm][sn] = wmma_f16(af[sm], bf[sn], acc[sm][sn]);

    // ---- commit B-tile of k+1 to LDS
    if (have_next) {
      half_t* bp = &Bs[nxt][brow * LDH + bko];
      if (knext < 32) {
        bp[0] = (half_t)((float)pc0.x * ps); bp[1] = (half_t)((float)pc0.y * ps);
        bp[2] = (half_t)((float)pc0.z * ps); bp[3] = (half_t)((float)pc0.w * ps);
        bp[4] = (half_t)((float)pc1.x * ps); bp[5] = (half_t)((float)pc1.y * ps);
        bp[6] = (half_t)((float)pc1.z * ps); bp[7] = (half_t)((float)pc1.w * ps);
      } else {
        bp[0] = (half_t)pf0.x; bp[1] = (half_t)pf0.y;
        bp[2] = (half_t)pf0.z; bp[3] = (half_t)pf0.w;
        bp[4] = (half_t)pf1.x; bp[5] = (half_t)pf1.y;
        bp[6] = (half_t)pf1.z; bp[7] = (half_t)pf1.w;
      }
    }
    wait_asynccnt0();
    __syncthreads();
  }

  // Epilogue: + bias, optional ReLU; write f32 result and (optionally) f16 act
  const int nlo = lane & 15, hs = lane >> 4;
#pragma unroll
  for (int sm = 0; sm < 2; ++sm) {
#pragma unroll
    for (int sn = 0; sn < 2; ++sn) {
      const int col = bn + wn * 32 + sn * 16 + nlo;
      const float bv = bias[col];
#pragma unroll
      for (int r = 0; r < 8; ++r) {
        const int row = bm + wm * 32 + sm * 16 + r + 8 * hs;
        float v = acc[sm][sn][r] + bv;
        if (relu) v = fmaxf(v, 0.0f);
        out[(size_t)row * D_DIM + col] = v;
        if (act_out) act_out[(size_t)row * D_DIM + col] = (half_t)v;
      }
    }
  }
}

// ---------------------------------------------------------------------------
// h = r + h ; if (apply_ln) h = LN(h)*gamma+beta ; act = f16(h)
// One block per row; 256 threads x float4.
__global__ __launch_bounds__(256) void k_resln(
    const float* __restrict__ r, float* __restrict__ h,
    const float* __restrict__ g, const float* __restrict__ b, int apply_ln,
    half_t* __restrict__ act) {
  __shared__ float ssum[256], ssq[256];
  const int row = blockIdx.x, tid = threadIdx.x;
  const float* rp = r + (size_t)row * D_DIM;
  float* hp = h + (size_t)row * D_DIM;
  float4 rv = ((const float4*)rp)[tid];
  float4 hv = ((const float4*)hp)[tid];
  float v0 = rv.x + hv.x, v1 = rv.y + hv.y, v2 = rv.z + hv.z, v3 = rv.w + hv.w;
  ssum[tid] = v0 + v1 + v2 + v3;
  ssq[tid]  = v0 * v0 + v1 * v1 + v2 * v2 + v3 * v3;
  __syncthreads();
  for (int s = 128; s > 0; s >>= 1) {
    if (tid < s) { ssum[tid] += ssum[tid + s]; ssq[tid] += ssq[tid + s]; }
    __syncthreads();
  }
  const float mean = ssum[0] * (1.0f / D_DIM);
  const float var  = ssq[0] * (1.0f / D_DIM) - mean * mean;
  const float rstd = rsqrtf(var + 1e-5f);
  const int k = tid * 4;
  float o0, o1, o2, o3;
  if (apply_ln) {
    o0 = (v0 - mean) * rstd * g[k + 0] + b[k + 0];
    o1 = (v1 - mean) * rstd * g[k + 1] + b[k + 1];
    o2 = (v2 - mean) * rstd * g[k + 2] + b[k + 2];
    o3 = (v3 - mean) * rstd * g[k + 3] + b[k + 3];
  } else {
    o0 = v0; o1 = v1; o2 = v2; o3 = v3;
  }
  float4 ov; ov.x = o0; ov.y = o1; ov.z = o2; ov.w = o3;
  ((float4*)hp)[tid] = ov;
  v4h av;
  av[0] = (half_t)o0; av[1] = (half_t)o1; av[2] = (half_t)o2; av[3] = (half_t)o3;
  *(v4h*)&act[(size_t)row * D_DIM + k] = av;
}

// ---------------------------------------------------------------------------
extern "C" void kernel_launch(void* const* d_in, const int* in_sizes, int n_in,
                              void* d_out, int out_size, void* d_ws, size_t ws_size,
                              hipStream_t stream) {
  const float* x      = (const float*)d_in[0];
  const int*   codes  = (const int*)d_in[1];
  const float* scales = (const float*)d_in[2];
  const float* bias   = (const float*)d_in[3];
  const float* la     = (const float*)d_in[4];
  const float* lb     = (const float*)d_in[5];
  const float* gamma  = (const float*)d_in[6];
  const float* beta   = (const float*)d_in[7];

  float* h = (float*)d_out;  // residual stream lives in d_out (f32 [1024,1024])
  char* base = (char*)d_ws;
  half_t* actA = (half_t*)(base);                 // 2 MB  f16 activations (ping)
  half_t* actB = (half_t*)(base + (2u << 20));    // 2 MB  f16 activations (pong)
  float*  rbuf = (float*)(base + (4u << 20));     // 4 MB  f32 linear output
  half_t* tbuf = (half_t*)(base + (8u << 20));    // 64 KB f16 ALPHA*(h@a^T)
  half_t* cur = actA;
  half_t* nxt = actB;

  k_init<<<1024, 256, 0, stream>>>(x, h, cur);

  for (int i = 0; i < NBLK; ++i) {
    for (int j = 0; j < NLIN; ++j) {
      const int lin = i * NLIN + j;
      k_lora_t<<<8, 256, 0, stream>>>(cur, la + (size_t)lin * R_DIM * D_DIM, tbuf);
      const bool last = (j == NLIN - 1);
      k_gemm<<<dim3(16, 8), 256, 0, stream>>>(
          cur,
          codes  + (size_t)lin * D_DIM * D_DIM,
          scales + (size_t)lin * D_DIM * (D_DIM / G_DIM),
          bias   + (size_t)lin * D_DIM,
          lb     + (size_t)lin * D_DIM * R_DIM,
          tbuf, rbuf,
          last ? (half_t*)nullptr : nxt,
          last ? 0 : 1);
      if (!last) { half_t* t = cur; cur = nxt; nxt = t; }
    }
    k_resln<<<1024, 256, 0, stream>>>(rbuf, h,
                                      gamma + (size_t)i * D_DIM,
                                      beta  + (size_t)i * D_DIM,
                                      (i < NBLK - 1) ? 1 : 0, cur);
  }
  (void)in_sizes; (void)n_in; (void)out_size; (void)ws_size;
}
